// AttentionGraphModel_27436251086855
// MI455X (gfx1250) — matrix operand order repaired and checked by reading, not compiled, not run on hardware
//
#include <hip/hip_runtime.h>
#include <hip/hip_bf16.h>

#define NN    20000
#define DEG   16
#define IN_F  512
#define HID   256

typedef __attribute__((ext_vector_type(16))) _Float16 v16h;
typedef __attribute__((ext_vector_type(8)))  _Float16 v8h;
typedef __attribute__((ext_vector_type(8)))  float    v8f;

__global__ __launch_bounds__(256)
void k_f32_to_f16(const float* __restrict__ in, _Float16* __restrict__ out, int n) {
    int i = blockIdx.x * 256 + threadIdx.x;
    if (i < n) out[i] = (_Float16)in[i];
}

// W: (K,N) f32 row-major  ->  Wt: (N,K) f16 (each output column contiguous)
__global__ __launch_bounds__(256)
void k_transpose_w(const float* __restrict__ W, _Float16* __restrict__ Wt, int K, int N) {
    int i = blockIdx.x * 256 + threadIdx.x;
    if (i < K * N) {
        int n = i / K, k = i - n * K;
        Wt[i] = (_Float16)W[(size_t)k * N + n];
    }
}

// C(M,256) = leaky_relu( A(M,K:f16) @ W(K,256) + bias, 0.2 )
// Wt is (256,K) f16. One block = 4 waves, each wave: 16x64 tile via wmma f16.
__global__ __launch_bounds__(128)
void k_gemm_bias_lrelu(const _Float16* __restrict__ A,
                       const _Float16* __restrict__ Bt,
                       const float* __restrict__ bias,
                       float* __restrict__ C, int K) {
    const int wave = threadIdx.x >> 5;
    const int lane = threadIdx.x & 31;
    const int l15  = lane & 15;
    const int hsel = lane >> 4;            // 0 or 1
    const int m0   = blockIdx.x * 16;
    const int n0   = wave * 64;
    const int N    = 256;

    // A fragment, contiguous k order chosen here (16x32): lanes 0-15 row=lane, K=0..15;
    // lanes 16-31 row=lane-16, K=16..31 (matches B).
    const _Float16* aptr = A + (size_t)(m0 + l15) * K + hsel * 16;

    v8f acc[4] = {};

    for (int k0 = 0; k0 < K; k0 += 32) {
        v16h a = *(const v16h*)(aptr + k0);
        #pragma unroll
        for (int s = 0; s < 4; ++s) {
            // B fragment, same contiguous k order as A (32x16): lanes 0-15 col=lane, K=k0..k0+15;
            // lanes 16-31 col=lane-16, K=k0+16..k0+31 -> contiguous 32B per lane.
            const int col = n0 + s * 16 + l15;
            v16h b = *(const v16h*)(Bt + (size_t)col * K + k0 + hsel * 16);
            acc[s] = __builtin_amdgcn_wmma_f32_16x16x32_f16(
                false, a, false, b, (short)0, acc[s], false, false);
        }
    }

    // C/D layout: VGPR v -> row = v + 8*hsel, col = lane&15 (+tile offset)
    #pragma unroll
    for (int s = 0; s < 4; ++s) {
        const int col = n0 + s * 16 + l15;
        const float bv = bias[col];
        #pragma unroll
        for (int v = 0; v < 8; ++v) {
            const int row = m0 + v + hsel * 8;
            float val = acc[s][v] + bv;
            val = val > 0.f ? val : 0.2f * val;
            C[(size_t)row * N + col] = val;
        }
    }
}

// Per-node attention scores for heads 2,3 only (heads 0,1 are src-constant ->
// uniform softmax; head bias c cancels in softmax).  s[i][p] = h[i] . (A[p,:F]+A[p,F:])
__global__ __launch_bounds__(256)
void k_att_scores(const float* __restrict__ h, const float* __restrict__ A,
                  float* __restrict__ s, int F) {
    const int wave = threadIdx.x >> 5, lane = threadIdx.x & 31;
    const int node = blockIdx.x * 8 + wave;
    if (node >= NN) return;
    const float* hr = h + (size_t)node * F;
    const int twoF = 2 * F;
    float a2 = 0.f, a3 = 0.f;
    for (int f = lane; f < F; f += 32) {
        const float hv = hr[f];
        a2 += hv * (A[2 * twoF + f] + A[2 * twoF + F + f]);
        a3 += hv * (A[3 * twoF + f] + A[3 * twoF + F + f]);
    }
    #pragma unroll
    for (int off = 16; off > 0; off >>= 1) {
        a2 += __shfl_down(a2, off, 32);
        a3 += __shfl_down(a3, off, 32);
    }
    if (lane == 0) { s[node * 2 + 0] = a2; s[node * 2 + 1] = a3; }
}

// One block (256 thr) per node: softmax over its 16 neighbors (heads 2,3),
// uniform 1/16 for heads 0,1 (which are identical), gather-accumulate, ELU.
// Non-final: write f16 interleaved (N, F*4) [idx f*4+p] for next GEMM.
// Final: write f32 mean-over-heads ELU to outf.
__global__ __launch_bounds__(256)
void k_aggregate(const float* __restrict__ h, const float* __restrict__ s,
                 const int* __restrict__ dst, _Float16* __restrict__ y16,
                 float* __restrict__ outf, int F, int final_layer) {
    __shared__ int   dS[DEG];
    __shared__ float aS[DEG][2];
    const int node = blockIdx.x;
    const int tid  = threadIdx.x;

    if (tid < DEG) {
        const int d = dst[node * DEG + tid];
        dS[tid] = d;
        aS[tid][0] = s[d * 2 + 0];
        aS[tid][1] = s[d * 2 + 1];
    }
    __syncthreads();

    float alpha = 0.f;
    const int j = tid & 15, head = (tid >> 4) & 1;
    if (tid < 32) {
        float mx = -1e30f;
        #pragma unroll
        for (int i = 0; i < DEG; ++i) mx = fmaxf(mx, aS[i][head]);
        float sm = 0.f;
        #pragma unroll
        for (int i = 0; i < DEG; ++i) sm += __expf(aS[i][head] - mx);
        alpha = __expf(aS[j][head] - mx) / sm;
    }
    __syncthreads();
    if (tid < 32) aS[j][head] = alpha;
    __syncthreads();

    const int f = tid;  // blockDim == F == 256
    float aU = 0.f, a2 = 0.f, a3 = 0.f;
    #pragma unroll
    for (int k = 0; k < DEG; ++k) {
        const float v = h[(size_t)dS[k] * F + f];
        aU += v;
        a2 += aS[k][0] * v;
        a3 += aS[k][1] * v;
    }
    aU *= (1.0f / DEG);

    if (final_layer) {
        float m = 0.25f * (aU + aU + a2 + a3);
        m = m > 0.f ? m : (__expf(m) - 1.f);
        outf[(size_t)node * F + f] = m;
    } else {
        float e0 = aU > 0.f ? aU : (__expf(aU) - 1.f);
        float e2 = a2 > 0.f ? a2 : (__expf(a2) - 1.f);
        float e3 = a3 > 0.f ? a3 : (__expf(a3) - 1.f);
        const size_t base = (size_t)node * F * 4 + (size_t)f * 4;
        y16[base + 0] = (_Float16)e0;
        y16[base + 1] = (_Float16)e0;   // head 1 == head 0
        y16[base + 2] = (_Float16)e2;
        y16[base + 3] = (_Float16)e3;
    }
}

extern "C" void kernel_launch(void* const* d_in, const int* in_sizes, int n_in,
                              void* d_out, int out_size, void* d_ws, size_t ws_size,
                              hipStream_t stream) {
    (void)in_sizes; (void)n_in; (void)out_size; (void)ws_size;

    const float* x  = (const float*)d_in[0];
    const float* W0 = (const float*)d_in[1];
    const float* b0 = (const float*)d_in[2];
    const float* A0 = (const float*)d_in[3];
    const float* W1 = (const float*)d_in[5];
    const float* b1 = (const float*)d_in[6];
    const float* A1 = (const float*)d_in[7];
    const float* W2 = (const float*)d_in[9];
    const float* b2 = (const float*)d_in[10];
    const float* A2 = (const float*)d_in[11];
    const int*   ei = (const int*)d_in[13];

    const int E = NN * DEG;
    const int* dst = ei + E;   // edge_index row 1

    char* ws = (char*)d_ws;
    auto alloc = [&](size_t bytes) {
        char* p = ws;
        ws += (bytes + 255) & ~(size_t)255;
        return p;
    };
    _Float16* x16 = (_Float16*)alloc((size_t)NN * IN_F * sizeof(_Float16));
    _Float16* w0t = (_Float16*)alloc((size_t)HID * IN_F * sizeof(_Float16));
    _Float16* w1t = (_Float16*)alloc((size_t)HID * 4 * HID * sizeof(_Float16));
    _Float16* w2t = (_Float16*)alloc((size_t)HID * 4 * HID * sizeof(_Float16));
    float*    hb  = (float*)   alloc((size_t)NN * HID * sizeof(float));
    float*    sb  = (float*)   alloc((size_t)NN * 2 * sizeof(float));
    _Float16* y16 = (_Float16*)alloc((size_t)NN * 4 * HID * sizeof(_Float16));
    float*    out = (float*)d_out;

    // Prep: fp16 copy of x, N-major fp16 weights
    k_f32_to_f16<<<(NN * IN_F + 255) / 256, 256, 0, stream>>>(x, x16, NN * IN_F);
    k_transpose_w<<<(IN_F * HID + 255) / 256, 256, 0, stream>>>(W0, w0t, IN_F, HID);
    k_transpose_w<<<(4 * HID * HID + 255) / 256, 256, 0, stream>>>(W1, w1t, 4 * HID, HID);
    k_transpose_w<<<(4 * HID * HID + 255) / 256, 256, 0, stream>>>(W2, w2t, 4 * HID, HID);

    // Layer 0
    k_gemm_bias_lrelu<<<NN / 16, 128, 0, stream>>>(x16, w0t, b0, hb, IN_F);
    k_att_scores<<<(NN + 7) / 8, 256, 0, stream>>>(hb, A0, sb, HID);
    k_aggregate<<<NN, 256, 0, stream>>>(hb, sb, dst, y16, nullptr, HID, 0);

    // Layer 1
    k_gemm_bias_lrelu<<<NN / 16, 128, 0, stream>>>(y16, w1t, b1, hb, 4 * HID);
    k_att_scores<<<(NN + 7) / 8, 256, 0, stream>>>(hb, A1, sb, HID);
    k_aggregate<<<NN, 256, 0, stream>>>(hb, sb, dst, y16, nullptr, HID, 0);

    // Layer 2 (final)
    k_gemm_bias_lrelu<<<NN / 16, 128, 0, stream>>>(y16, w2t, b2, hb, 4 * HID);
    k_att_scores<<<(NN + 7) / 8, 256, 0, stream>>>(hb, A2, sb, HID);
    k_aggregate<<<NN, 256, 0, stream>>>(hb, sb, dst, nullptr, out, HID, 1);
}
